// PagedAttentionImpl_30983894073769
// MI455X (gfx1250) — matrix operand order repaired and checked by reading, not compiled, not run on hardware
//
#include <hip/hip_runtime.h>

#define S_      16
#define HQ_     32
#define HK_     8
#define D_      128
#define GQ_     4        // HQ/HK
#define BS_     16
#define BPS_    256
#define NCHUNK_ 16
#define CHUNK_  256
#define WPB_    4        // waves per block in the attention kernel
#define EPS_    1e-6f
#define SCALE_  0.08838834764831845f   // 1/sqrt(128)

typedef __attribute__((ext_vector_type(16))) __bf16 v16bf;
typedef __attribute__((ext_vector_type(8)))  float  v8f;

// ---------------------------------------------------------------------------
// Kernel 1: fused RMSNorm + NeoX RoPE for q (-> bf16) and k (-> f32 "k_new").
// One 128-thread block per head instance. grid = S*(HQ+HK).
// ---------------------------------------------------------------------------
__global__ __launch_bounds__(128) void qk_norm_rope_kernel(
    const float* __restrict__ q, const float* __restrict__ k,
    const float* __restrict__ qw, const float* __restrict__ kw,
    const float* __restrict__ cs_cache, const int* __restrict__ position,
    __bf16* __restrict__ qbf, float* __restrict__ knew)
{
  const int bid = blockIdx.x;
  const int d   = threadIdx.x;               // 0..127
  const bool isQ = bid < S_ * HQ_;
  int s, h;
  const float* x;
  const float* w;
  if (isQ) { s = bid / HQ_; h = bid % HQ_; x = q + (size_t)(s * HQ_ + h) * D_; w = qw; }
  else { int b2 = bid - S_ * HQ_; s = b2 / HK_; h = b2 % HK_; x = k + (size_t)(s * HK_ + h) * D_; w = kw; }

  const float v = x[d];
  __shared__ float red[D_];
  red[d] = v * v;
  __syncthreads();
  for (int off = D_ / 2; off > 0; off >>= 1) {
    if (d < off) red[d] += red[d + off];
    __syncthreads();
  }
  const float inv = rsqrtf(red[0] * (1.0f / D_) + EPS_);
  __syncthreads();
  const float xn = v * inv * w[d];
  red[d] = xn;
  __syncthreads();

  const float* cs = cs_cache + (size_t)position[s] * D_;   // [cos(64) | sin(64)]
  float o;
  if (d < D_ / 2) {
    o = xn * cs[d] - red[d + D_ / 2] * cs[D_ / 2 + d];
  } else {
    const int d2 = d - D_ / 2;
    o = xn * cs[d2] + red[d2] * cs[D_ / 2 + d2];
  }
  if (isQ) qbf[(size_t)(s * HQ_ + h) * D_ + d] = (__bf16)o;
  else     knew[(size_t)(s * HK_ + h) * D_ + d] = o;
}

// ---------------------------------------------------------------------------
// Kernel 2: flash-decoding chunk kernel.
// grid = S*HK*(NCHUNK/WPB), block = 32*WPB. Each wave owns one 256-token
// context chunk of one (seq, kv-head) and emits partial (m, l, unnormalized
// acc). 2048 independent waves total.
// ---------------------------------------------------------------------------
__global__ __launch_bounds__(32 * WPB_) void paged_attn_chunk_kernel(
    const float* __restrict__ kc, const float* __restrict__ vc,
    const float* __restrict__ knew, const float* __restrict__ vin,
    const __bf16* __restrict__ qbf,
    const int* __restrict__ position, const int* __restrict__ block_tables,
    const int* __restrict__ context_lens,
    float* __restrict__ m_ws, float* __restrict__ l_ws, float* __restrict__ a_ws)
{
  const int sh    = blockIdx.x % (S_ * HK_);
  const int quad  = blockIdx.x / (S_ * HK_);
  const int s     = sh / HK_;
  const int hk    = sh % HK_;
  const int wave  = threadIdx.x >> 5;
  const int lane  = threadIdx.x & 31;
  const int mrow  = lane & 15;      // C-layout column / A-layout row / B-layout column
  const int hi    = lane >> 4;      // lane half
  const int koff  = hi << 3;        // A-layout K offset pattern

  const int chunk = quad * WPB_ + wave;
  const int ctx   = context_lens[s];
  const int pos   = position[s];
  const int base  = chunk * CHUNK_;
  const int idxML = (sh * NCHUNK_ + chunk) * GQ_;

  __shared__ __bf16 plds[WPB_][GQ_][32];

  if (base >= ctx) {
    if (lane == 0) {
      #pragma unroll
      for (int g = 0; g < GQ_; ++g) { m_ws[idxML + g] = -1e30f; l_ws[idxML + g] = 0.0f; }
    }
    return;
  }
  const int nkeys = min(CHUNK_, ctx - base);

  // ---- Q A-fragments: 4 K-steps of 32 dims each. Contiguous k order here (as in B): lane half low
  // holds K {0..15}, high holds {16..31} within each step. ----
  v16bf aq[4];
  const __bf16* qp = qbf + (size_t)(s * HQ_ + hk * GQ_ + mrow) * D_;
  #pragma unroll
  for (int st = 0; st < 4; ++st) {
    #pragma unroll
    for (int e = 0; e < 16; ++e) {
      const int dd = st * 32 + e + (koff << 1);
      aq[st][e] = (mrow < GQ_) ? qp[dd] : (__bf16)0.0f;
    }
  }

  float mst[GQ_], lst[GQ_];
  #pragma unroll
  for (int g = 0; g < GQ_; ++g) { mst[g] = -1e30f; lst[g] = 0.0f; }
  v8f acc[8];
  #pragma unroll
  for (int dt = 0; dt < 8; ++dt) {
    #pragma unroll
    for (int e = 0; e < 8; ++e) acc[dt][e] = 0.0f;
  }

  const int*   btrow    = block_tables + s * BPS_;
  const float* knew_row = knew + (size_t)(s * HK_ + hk) * D_;
  const float* vnew_row = vin  + (size_t)(s * HK_ + hk) * D_;

  for (int t0 = base; t0 < base + nkeys; t0 += 32) {
    // ---- prefetch next group's K/V rows (global_prefetch_b8; no counters) ----
    if (t0 + 32 < base + nkeys) {
      #pragma unroll
      for (int j = 0; j < 2; ++j) {
        const int tn  = t0 + 32 + j * 16;
        const int btn = __builtin_amdgcn_readfirstlane(btrow[tn >> 4]);
        const size_t row = (((size_t)btn * BS_ + mrow) * HK_ + hk) * D_;
        __builtin_prefetch((const void*)(kc + row + (hi << 4)), 0, 3);
        __builtin_prefetch((const void*)(vc + row + (hi << 6)), 0, 3);
      }
    }

    // ---------------- QK^T over two 16-key subtiles ----------------
    v8f sc[2];
    int bt[2];
    #pragma unroll
    for (int j = 0; j < 2; ++j) {
      const int tile = t0 + j * 16;               // tile aligned to one cache block
      bt[j] = __builtin_amdgcn_readfirstlane(btrow[tile >> 4]);
      const int tkey = tile + mrow;
      const float* kp = (tkey == pos) ? knew_row
                      : (kc + (((size_t)bt[j] * BS_ + mrow) * HK_ + hk) * D_);
      v8f sa;
      #pragma unroll
      for (int e = 0; e < 8; ++e) sa[e] = 0.0f;
      #pragma unroll
      for (int st = 0; st < 4; ++st) {
        // B layout: lane = key column, element e -> K = e + 16*hi (contiguous dims)
        const int d0 = st * 32 + (hi << 4);
        const float4 f0 = *(const float4*)(kp + d0);
        const float4 f1 = *(const float4*)(kp + d0 + 4);
        const float4 f2 = *(const float4*)(kp + d0 + 8);
        const float4 f3 = *(const float4*)(kp + d0 + 12);
        v16bf bk;
        bk[0]=(__bf16)f0.x;  bk[1]=(__bf16)f0.y;  bk[2]=(__bf16)f0.z;  bk[3]=(__bf16)f0.w;
        bk[4]=(__bf16)f1.x;  bk[5]=(__bf16)f1.y;  bk[6]=(__bf16)f1.z;  bk[7]=(__bf16)f1.w;
        bk[8]=(__bf16)f2.x;  bk[9]=(__bf16)f2.y;  bk[10]=(__bf16)f2.z; bk[11]=(__bf16)f2.w;
        bk[12]=(__bf16)f3.x; bk[13]=(__bf16)f3.y; bk[14]=(__bf16)f3.z; bk[15]=(__bf16)f3.w;
        sa = __builtin_amdgcn_wmma_f32_16x16x32_bf16(false, aq[st], false, bk,
                                                     (short)0, sa, false, false);
      }
      sc[j] = sa;
    }

    // ---------------- online softmax for this 32-key group ----------------
    const bool v0 = (t0 + mrow) < ctx;
    const bool v1 = (t0 + 16 + mrow) < ctx;
    float p0[GQ_], p1[GQ_], alpha[GQ_];
    #pragma unroll
    for (int g = 0; g < GQ_; ++g) {
      float s0 = v0 ? sc[0][g] * SCALE_ : -1e30f;
      float s1 = v1 ? sc[1][g] * SCALE_ : -1e30f;
      float rm = fmaxf(s0, s1);
      #pragma unroll
      for (int off = 8; off >= 1; off >>= 1)
        rm = fmaxf(rm, __shfl_xor(rm, off, 32));   // reduces within 16-lane group
      const float mnew = fmaxf(mst[g], rm);
      alpha[g] = __expf(mst[g] - mnew);
      p0[g] = v0 ? __expf(s0 - mnew) : 0.0f;
      p1[g] = v1 ? __expf(s1 - mnew) : 0.0f;
      float rs = p0[g] + p1[g];
      #pragma unroll
      for (int off = 8; off >= 1; off >>= 1)
        rs += __shfl_xor(rs, off, 32);
      lst[g] = lst[g] * alpha[g] + rs;
      mst[g] = mnew;
    }

    // probs C-layout -> LDS (rows 0..3 only; rows >=4 fragment-zeroed below)
    if (hi == 0) {
      #pragma unroll
      for (int g = 0; g < GQ_; ++g) {
        plds[wave][g][mrow]      = (__bf16)p0[g];
        plds[wave][g][16 + mrow] = (__bf16)p1[g];
      }
    }
    asm volatile("s_wait_dscnt 0" ::: "memory");   // single-wave DS in-order + wait

    // probs A-fragment (16x32)
    v16bf ap;
    #pragma unroll
    for (int e = 0; e < 16; ++e) {
      const int kl = e + (koff << 1);
      ap[e] = (mrow < GQ_) ? plds[wave][mrow][kl] : (__bf16)0.0f;
    }
    asm volatile("" ::: "memory");

    // ---------------- PV: acc += P(16x32) x V(32 x 16-dim tile) ----------------
    #pragma unroll
    for (int dt = 0; dt < 8; ++dt) {
      v16bf bv;
      #pragma unroll
      for (int e = 0; e < 16; ++e) {
        const int kl  = e + (hi << 4);     // B-layout K = key within 32-group
        const int t   = t0 + kl;
        const int btv = hi ? bt[1] : bt[0];
        const float* vp = (t == pos) ? vnew_row
                        : (vc + (((size_t)btv * BS_ + e) * HK_ + hk) * D_);
        bv[e] = (__bf16)vp[dt * 16 + mrow];
      }
      #pragma unroll
      for (int e = 0; e < 8; ++e) acc[dt][e] *= alpha[e & 3];
      acc[dt] = __builtin_amdgcn_wmma_f32_16x16x32_bf16(false, ap, false, bv,
                                                        (short)0, acc[dt], false, false);
    }
  }

  // ---------------- write partials ----------------
  if (lane == 0) {
    #pragma unroll
    for (int g = 0; g < GQ_; ++g) { m_ws[idxML + g] = mst[g]; l_ws[idxML + g] = lst[g]; }
  }
  if (hi == 0) {
    #pragma unroll
    for (int g = 0; g < GQ_; ++g) {
      #pragma unroll
      for (int dt = 0; dt < 8; ++dt)
        a_ws[(size_t)(idxML + g) * D_ + dt * 16 + mrow] = acc[dt][g];
    }
  }
}

// ---------------------------------------------------------------------------
// Kernel 3: flash-decoding combine over NCHUNK partials. grid = S*HK, 128 thr.
// ---------------------------------------------------------------------------
__global__ __launch_bounds__(128) void combine_kernel(
    const float* __restrict__ m_ws, const float* __restrict__ l_ws,
    const float* __restrict__ a_ws, float* __restrict__ out)
{
  const int sh = blockIdx.x;
  const int s  = sh / HK_;
  const int hk = sh % HK_;
  const int d  = threadIdx.x;
  for (int g = 0; g < GQ_; ++g) {
    float M = -1e30f;
    #pragma unroll
    for (int w = 0; w < NCHUNK_; ++w)
      M = fmaxf(M, m_ws[(sh * NCHUNK_ + w) * GQ_ + g]);
    float num = 0.0f, den = 0.0f;
    #pragma unroll
    for (int w = 0; w < NCHUNK_; ++w) {
      const int i = (sh * NCHUNK_ + w) * GQ_ + g;
      const float lw = l_ws[i];
      if (lw > 0.0f) {                 // skip empty chunks (acc never written)
        const float wt = __expf(m_ws[i] - M);
        den += wt * lw;
        num += wt * a_ws[(size_t)i * D_ + d];
      }
    }
    out[(size_t)(s * HQ_ + hk * GQ_ + g) * D_ + d] = num / den;
  }
}

// ---------------------------------------------------------------------------
extern "C" void kernel_launch(void* const* d_in, const int* in_sizes, int n_in,
                              void* d_out, int out_size, void* d_ws, size_t ws_size,
                              hipStream_t stream) {
  const float* q  = (const float*)d_in[0];
  const float* k  = (const float*)d_in[1];
  const float* v  = (const float*)d_in[2];
  const float* qw = (const float*)d_in[3];
  const float* kw = (const float*)d_in[4];
  const float* cs = (const float*)d_in[5];
  const float* kc = (const float*)d_in[6];
  const float* vc = (const float*)d_in[7];
  const int* position     = (const int*)d_in[8];
  // d_in[9] slot_mapping: implied by block_tables/position for this layout
  const int* block_tables = (const int*)d_in[10];
  const int* context_lens = (const int*)d_in[11];
  float* out = (float*)d_out;

  char* ws = (char*)d_ws;
  __bf16* qbf  = (__bf16*)ws;                  // S*HQ*D bf16          = 128 KB
  float*  knew = (float*)(ws + 16 * 1024);     // S*HK*D f32           = 64 KB
  float*  m_ws = (float*)(ws + 80 * 1024);     // S*HK*NCHUNK*GQ f32   = 32 KB
  float*  l_ws = (float*)(ws + 112 * 1024);    // S*HK*NCHUNK*GQ f32   = 32 KB
  float*  a_ws = (float*)(ws + 144 * 1024);    // ... * D f32          = 4 MB

  qk_norm_rope_kernel<<<S_ * (HQ_ + HK_), 128, 0, stream>>>(
      q, k, qw, kw, cs, position, qbf, knew);
  paged_attn_chunk_kernel<<<S_ * HK_ * (NCHUNK_ / WPB_), 32 * WPB_, 0, stream>>>(
      kc, vc, knew, v, qbf, position, block_tables, context_lens, m_ws, l_ws, a_ws);
  combine_kernel<<<S_ * HK_, 128, 0, stream>>>(m_ws, l_ws, a_ws, out);
}
